// GraphRespiratory_75788992905558
// MI455X (gfx1250) — matrix operand order repaired and checked by reading, not compiled, not run on hardware
//
#include <hip/hip_runtime.h>
#include <hip/hip_bf16.h>
#include <math.h>

// ---------------------------------------------------------------------------
// Sizes from the reference: N=4096, D=128, C=5, K=5, M=16, B=100, L=3
// ---------------------------------------------------------------------------
#define NN 4096
#define DD 128
#define CC 5
#define MM 16
#define BB 100
#define LL 3

typedef __attribute__((ext_vector_type(16))) _Float16 v16h;
typedef __attribute__((ext_vector_type(8)))  float    v8f;

// ---------------------------------------------------------------------------
// Workspace layout (bytes). Total ~11 MB.
// ---------------------------------------------------------------------------
constexpr size_t OFF_X    = 0;                               // f32  N*D
constexpr size_t OFF_XH   = OFF_X    + (size_t)NN*DD*4;      // f16  N*D
constexpr size_t OFF_EMBH = OFF_XH   + (size_t)NN*DD*2;      // f16  N*D
constexpr size_t OFF_GI   = OFF_EMBH + (size_t)NN*DD*2;      // f32  N*3D
constexpr size_t OFF_S    = OFF_GI   + (size_t)NN*3*DD*4;    // f32  M*N   (attn logits)
constexpr size_t OFF_SH   = OFF_S    + (size_t)MM*NN*4;      // f16  M*N   (attn probs)
constexpr size_t OFF_ABN  = OFF_SH   + (size_t)MM*NN*2;      // f32  N
constexpr size_t OFF_VAB  = OFF_ABN  + (size_t)NN*4;         // f32  N
constexpr size_t OFF_CMB  = OFF_VAB  + (size_t)NN*4;         // f32  N
constexpr size_t OFF_QF   = OFF_CMB  + (size_t)NN*4;         // f32  M*D
constexpr size_t OFF_ST   = OFF_QF   + (size_t)MM*DD*4;      // f32  M
constexpr size_t OFF_EN   = OFF_ST   + 64;                   // f32  M
constexpr size_t OFF_GG   = OFF_EN   + 64;                   // f32  2 (ga,gv)
constexpr size_t OFF_DIV  = OFF_GG   + 64;                   // f32  1
constexpr size_t OFF_HFF  = OFF_DIV  + 64;                   // f32  M*D  (final hf)
constexpr size_t OFF_LA   = OFF_HFF  + (size_t)MM*DD*4;      // f32  M    (final local_ab)
constexpr size_t OFF_LOG  = OFF_LA   + 64;                   // f32  L*M*2B refinement logits

// ---------------------------------------------------------------------------
// WMMA helpers (wave32, 16x16x32 f16 -> f32). Fragment layouts used here
// (contiguous k, not the ISA table order): A 16x32 (lane m=l%16, k-base 0/16 by half-wave, pairs packed per
// VGPR); B 32x16 (lane col=l%16, k-base 0/16 by half-wave); C/D 16x16
// (lane n=l%16, VGPR r -> M=r or r+8 by half-wave).
// ---------------------------------------------------------------------------
__device__ inline v8f wmma32(v16h a, v16h b, v8f c) {
  return __builtin_amdgcn_wmma_f32_16x16x32_f16(
      /*neg_a=*/false, a, /*neg_b=*/false, b,
      /*c_mod=*/(short)0, c, /*reuse_a=*/false, /*reuse_b=*/false);
}

template <typename F>
__device__ inline v16h make_A(F at, int lane) {
  int m  = lane & 15;
  int kb = (lane < 16) ? 0 : 16;
  v16h a;
#pragma unroll
  for (int h = 0; h < 8; ++h) {
    int k0 = kb + 2 * h;
    a[2 * h]     = at(m, k0);
    a[2 * h + 1] = at(m, k0 + 1);
  }
  return a;
}

template <typename F>
__device__ inline v16h make_B(F bt, int lane) {
  int c  = lane & 15;
  int kb = (lane < 16) ? 0 : 16;
  v16h b;
#pragma unroll
  for (int h = 0; h < 8; ++h) {
    int k0 = kb + 2 * h;
    b[2 * h]     = bt(k0, c);
    b[2 * h + 1] = bt(k0 + 1, c);
  }
  return b;
}

__device__ inline float sigm(float x) { return 1.f / (1.f + expf(-x)); }

__device__ inline float gru1(float h, float x, const float* wih, const float* whh,
                             const float* bih, const float* bhh) {
  float gi0 = wih[0] * x + bih[0], gi1 = wih[1] * x + bih[1], gi2 = wih[2] * x + bih[2];
  float gh0 = h * whh[0] + bhh[0], gh1 = h * whh[1] + bhh[1], gh2 = h * whh[2] + bhh[2];
  float r = sigm(gi0 + gh0);
  float z = sigm(gi1 + gh1);
  float n = tanhf(gi2 + r * gh2);
  return (1.f - z) * n + z * h;
}

// ---------------------------------------------------------------------------
// K1: per-node preprocessing. time feature MLP + layernorm, x = emb + tf,
// f16 mirrors for WMMA, depthwise smoothing conv + softmax -> abnormal,
// vad softmax, combined.
// ---------------------------------------------------------------------------
__global__ __launch_bounds__(256) void pre_kernel(
    const float* __restrict__ emb, const float* __restrict__ tpos,
    const float* __restrict__ pred, const float* __restrict__ vad,
    const float* __restrict__ W_te, const float* __restrict__ b_te,
    const float* __restrict__ ln_g, const float* __restrict__ ln_b,
    const float* __restrict__ gk,
    float* __restrict__ X, _Float16* __restrict__ Xh, _Float16* __restrict__ EMBh,
    float* __restrict__ abn, float* __restrict__ vab, float* __restrict__ cmb) {
  int i = blockIdx.x * blockDim.x + threadIdx.x;
  if (i >= NN) return;
  float t = tpos[i];
  float sum = 0.f, sq = 0.f;
  for (int j = 0; j < DD; ++j) {
    float h = W_te[2 * j] * t + W_te[2 * j + 1] + b_te[j];
    h = h > 0.f ? h : 0.f;
    sum += h; sq += h * h;
  }
  float mu   = sum * (1.f / DD);
  float var  = sq * (1.f / DD) - mu * mu;
  float rinv = rsqrtf(var + 1e-5f);
  for (int j = 0; j < DD; ++j) {
    float h = W_te[2 * j] * t + W_te[2 * j + 1] + b_te[j];
    h = h > 0.f ? h : 0.f;
    float tf = ln_g[j] * (h - mu) * rinv + ln_b[j];
    float e  = emb[i * DD + j];
    float xv = e + tf;
    X[i * DD + j]    = xv;
    Xh[i * DD + j]   = (_Float16)xv;
    EMBh[i * DD + j] = (_Float16)e;
  }
  // depthwise conv (zero pad) + softmax over C
  float s[CC];
  float mx = -1e30f;
  for (int c = 0; c < CC; ++c) {
    float acc = 0.f;
#pragma unroll
    for (int k = 0; k < 5; ++k) {
      int idx  = i + k - 2;
      float pv = (idx >= 0 && idx < NN) ? pred[idx * CC + c] : 0.f;
      acc += gk[c * 5 + k] * pv;
    }
    s[c] = acc;
    mx   = fmaxf(mx, acc);
  }
  float se = 0.f;
  for (int c = 0; c < CC; ++c) { s[c] = expf(s[c] - mx); se += s[c]; }
  float ab = 1.f - s[0] / se;
  float v0 = vad[i * 2], v1 = vad[i * 2 + 1];
  float vm = fmaxf(v0, v1);
  float e0 = expf(v0 - vm), e1 = expf(v1 - vm);
  float vv = e1 / (e0 + e1);
  abn[i] = ab;
  vab[i] = vv;
  cmb[i] = 0.5f * (ab + vv);
}

// ---------------------------------------------------------------------------
// K2: global scalar GRU scans (strictly sequential, trivial work).
// ---------------------------------------------------------------------------
__global__ void ggru_kernel(const float* __restrict__ abn, const float* __restrict__ vab,
                            const float* wia, const float* wha, const float* bia, const float* bha,
                            const float* wiv, const float* whv, const float* biv, const float* bhv,
                            float* __restrict__ gout) {
  int tid = threadIdx.x;
  if (tid == 0) {
    float h = 0.f;
    for (int t = 0; t < NN; ++t) h = gru1(h, abn[t], wia, wha, bia, bha);
    gout[0] = h;
  } else if (tid == 1) {
    float h = 0.f;
    for (int t = 0; t < NN; ++t) h = gru1(h, vab[t], wiv, whv, biv, bhv);
    gout[1] = h;
  }
}

// ---------------------------------------------------------------------------
// K3: attention logits S(16,4096) = queries @ emb^T via WMMA.
// One 16x16 output tile per wave; 256 tiles over 32 blocks x 8 waves.
// ---------------------------------------------------------------------------
__global__ __launch_bounds__(256) void attn_logits_kernel(
    const float* __restrict__ q, const _Float16* __restrict__ EMBh,
    float* __restrict__ S) {
  int lane = threadIdx.x & 31, wave = threadIdx.x >> 5;
  int tile = blockIdx.x * 8 + wave;  // 0..255
  int base = tile * 16;
  v8f c = {};
#pragma unroll
  for (int kc = 0; kc < 4; ++kc) {
    int k0 = kc * 32;
    v16h a = make_A([&](int m, int k) { return (_Float16)q[m * DD + k0 + k]; }, lane);
    v16h b = make_B([&](int k, int col) { return EMBh[(base + col) * DD + k0 + k]; }, lane);
    c = wmma32(a, b, c);
  }
  int n  = base + (lane & 15);
  int mb = (lane >= 16) ? 8 : 0;
#pragma unroll
  for (int r = 0; r < 8; ++r) S[(size_t)(r + mb) * NN + n] = c[r];
}

// ---------------------------------------------------------------------------
// K4: row softmax over 4096 (one block per row), emit f16 probs.
// ---------------------------------------------------------------------------
__global__ __launch_bounds__(256) void softmax_kernel(const float* __restrict__ S,
                                                      _Float16* __restrict__ Sh) {
  __shared__ float red[256];
  int row = blockIdx.x, tid = threadIdx.x;
  float mx = -1e30f;
  for (int j = tid; j < NN; j += 256) mx = fmaxf(mx, S[(size_t)row * NN + j]);
  red[tid] = mx;
  __syncthreads();
  for (int s = 128; s > 0; s >>= 1) { if (tid < s) red[tid] = fmaxf(red[tid], red[tid + s]); __syncthreads(); }
  mx = red[0];
  __syncthreads();
  float sm = 0.f;
  for (int j = tid; j < NN; j += 256) sm += expf(S[(size_t)row * NN + j] - mx);
  red[tid] = sm;
  __syncthreads();
  for (int s = 128; s > 0; s >>= 1) { if (tid < s) red[tid] += red[tid + s]; __syncthreads(); }
  float inv = 1.f / red[0];
  for (int j = tid; j < NN; j += 256)
    Sh[(size_t)row * NN + j] = (_Float16)(expf(S[(size_t)row * NN + j] - mx) * inv);
}

// ---------------------------------------------------------------------------
// K5: qf(16,128) = attn @ x, K=4096 reduction via chained WMMA. One block,
// 8 waves, each owns a 16-column tile of qf.
// ---------------------------------------------------------------------------
__global__ __launch_bounds__(256) void qf_kernel(const _Float16* __restrict__ Sh,
                                                 const _Float16* __restrict__ Xh,
                                                 float* __restrict__ qf) {
  int lane = threadIdx.x & 31, wave = threadIdx.x >> 5;
  int cb = wave * 16;
  v8f c = {};
  for (int kc = 0; kc < NN / 32; ++kc) {
    int k0 = kc * 32;
    v16h a = make_A([&](int m, int k) { return Sh[(size_t)m * NN + k0 + k]; }, lane);
    v16h b = make_B([&](int k, int col) { return Xh[(size_t)(k0 + k) * DD + cb + col]; }, lane);
    c = wmma32(a, b, c);
  }
  int n  = cb + (lane & 15);
  int mb = (lane >= 16) ? 8 : 0;
#pragma unroll
  for (int r = 0; r < 8; ++r) qf[(r + mb) * DD + n] = c[r];
}

// ---------------------------------------------------------------------------
// K6: diversity + interval head (tiny): qn, div, p = relu([qf|ga,gv]@Wg1^T)@Wg2^T,
// starts/ends.
// ---------------------------------------------------------------------------
__global__ __launch_bounds__(256) void interval_kernel(
    const float* __restrict__ qf, const float* __restrict__ gg,
    const float* __restrict__ Wg1, const float* __restrict__ bg1,
    const float* __restrict__ Wg2, const float* __restrict__ bg2,
    float* __restrict__ starts, float* __restrict__ ends, float* __restrict__ divout) {
  __shared__ float qn[MM][DD];
  __shared__ float h1[MM][DD];
  __shared__ float acc;
  int tid = threadIdx.x;
  if (tid == 0) acc = 0.f;
  if (tid < MM) {
    float s = 0.f;
    for (int k = 0; k < DD; ++k) { float v = qf[tid * DD + k]; s += v * v; }
    float inv = 1.f / fmaxf(sqrtf(s), 1e-8f);
    for (int k = 0; k < DD; ++k) qn[tid][k] = qf[tid * DD + k] * inv;
  }
  __syncthreads();
  if (tid < 120) {  // M*(M-1)/2 upper-triangle pairs
    int i = 0, t = tid;
    while (t >= 15 - i) { t -= 15 - i; ++i; }
    int j = i + 1 + t;
    float d = 0.f;
    for (int k = 0; k < DD; ++k) d += qn[i][k] * qn[j][k];
    atomicAdd(&acc, d);
  }
  __syncthreads();
  if (tid == 0) divout[0] = acc / 120.f;
  float ga = gg[0], gv = gg[1];
  for (int idx = tid; idx < MM * DD; idx += 256) {
    int m = idx >> 7, j = idx & 127;
    const float* wr = &Wg1[j * (DD + 2)];
    float s = bg1[j];
    for (int k = 0; k < DD; ++k) s += wr[k] * qf[m * DD + k];
    s += wr[DD] * ga + wr[DD + 1] * gv;
    h1[m][j] = s > 0.f ? s : 0.f;
  }
  __syncthreads();
  if (tid < MM) {
    int m = tid;
    float p0 = bg2[0], p1 = bg2[1];
    for (int j = 0; j < DD; ++j) { p0 += Wg2[j] * h1[m][j]; p1 += Wg2[DD + j] * h1[m][j]; }
    float cc = sigm(p0);
    float ww = sigm(p1) * 0.5f;
    starts[m] = fminf(fmaxf(cc - ww * 0.5f, 0.f), 1.f);
    ends[m]   = fminf(fmaxf(cc + ww * 0.5f, 0.f), 1.f);
  }
}

// ---------------------------------------------------------------------------
// K7: GI(4096,384) = X @ wih_lf^T + bih  (parallel WMMA GEMM; hoists the
// input half of the recurrent GRU out of the sequential scan).
// ---------------------------------------------------------------------------
__global__ __launch_bounds__(256) void gi_kernel(const _Float16* __restrict__ Xh,
                                                 const float* __restrict__ wih,
                                                 const float* __restrict__ bih,
                                                 float* __restrict__ GI) {
  int lane = threadIdx.x & 31, wave = threadIdx.x >> 5;
  int id = blockIdx.x * 8 + wave;  // 6144 tiles = 256 row-tiles x 24 col-tiles
  int rt = id / 24, ct = id % 24;
  int rbase = rt * 16, cbase = ct * 16;
  v8f c = {};
#pragma unroll
  for (int kc = 0; kc < 4; ++kc) {
    int k0 = kc * 32;
    v16h a = make_A([&](int m, int k) { return Xh[(size_t)(rbase + m) * DD + k0 + k]; }, lane);
    v16h b = make_B([&](int k, int col) { return (_Float16)wih[(size_t)(cbase + col) * DD + k0 + k]; }, lane);
    c = wmma32(a, b, c);
  }
  int n  = cbase + (lane & 15);
  int mb = (lane >= 16) ? 8 : 0;
  float bv = bih[n];
#pragma unroll
  for (int r = 0; r < 8; ++r) GI[(size_t)(rbase + r + mb) * (3 * DD) + n] = c[r] + bv;
}

// ---------------------------------------------------------------------------
// K8: the sequential GRU scan. One persistent workgroup (8 wave32). hf lives
// in LDS (f32 + f16 mirror). whh^T preloaded once as 12 WMMA B-fragments per
// wave. Each step: 96 v_wmma_f32_16x16x32_f16 across the WGP computes
// hf(16x128) @ whh^T(128x384); gates + masked update are lane-local because
// wave w owns matching r/z/n columns (tiles w, w+8, w+16).
// ---------------------------------------------------------------------------
__global__ __launch_bounds__(256) void scan_kernel(
    const float* __restrict__ GI, const float* __restrict__ whh,
    const float* __restrict__ bhh, const float* __restrict__ tpos,
    const float* __restrict__ cmb, const float* __restrict__ starts,
    const float* __restrict__ ends, const float* __restrict__ wila,
    const float* __restrict__ whla, const float* __restrict__ bila,
    const float* __restrict__ bhla, float* __restrict__ hf_out,
    float* __restrict__ la_out) {
  __shared__ float    hf[MM][DD];
  __shared__ _Float16 hfh[MM][DD];
  __shared__ float    st[MM], en[MM], la[MM];
  int tid = threadIdx.x, lane = tid & 31, wave = tid >> 5;
  for (int idx = tid; idx < MM * DD; idx += 256) {
    hf[idx >> 7][idx & 127]  = 0.f;
    hfh[idx >> 7][idx & 127] = (_Float16)0.f;
  }
  if (tid < MM) { st[tid] = starts[tid]; en[tid] = ends[tid]; la[tid] = 0.f; }
  __syncthreads();

  // Preload whh^T B-fragments: gates g in {r,z,n} -> output cols g*128 + wave*16 + c.
  v16h Bf[3][4];
  int colc = wave * 16;
#pragma unroll
  for (int g = 0; g < 3; ++g) {
#pragma unroll
    for (int kc = 0; kc < 4; ++kc) {
      int k0 = kc * 32;
      Bf[g][kc] = make_B(
          [&](int k, int col) { return (_Float16)whh[(size_t)(g * DD + colc + col) * DD + k0 + k]; },
          lane);
    }
  }
  int n  = lane & 15;
  int mb = (lane >= 16) ? 8 : 0;
  float bhr = bhh[0 * DD + colc + n];
  float bhz = bhh[1 * DD + colc + n];
  float bhn = bhh[2 * DD + colc + n];

  for (int t = 0; t < NN; ++t) {
    float tcur = tpos[t];
    __syncthreads();  // prior step's hf writes visible
    v16h A[4];
#pragma unroll
    for (int kc = 0; kc < 4; ++kc) {
      int k0 = kc * 32;
      A[kc] = make_A([&](int m, int k) { return hfh[m][k0 + k]; }, lane);
    }
    __syncthreads();  // all reads done before anyone writes
    v8f cr = {}, cz = {}, cn = {};
#pragma unroll
    for (int kc = 0; kc < 4; ++kc) {
      cr = wmma32(A[kc], Bf[0][kc], cr);
      cz = wmma32(A[kc], Bf[1][kc], cz);
      cn = wmma32(A[kc], Bf[2][kc], cn);
    }
    const float* gi = &GI[(size_t)t * (3 * DD)];
    float gir = gi[colc + n];
    float giz = gi[DD + colc + n];
    float gin = gi[2 * DD + colc + n];
    if (t + 1 < NN) {  // hide next GI row's latency behind the gate math
      __builtin_prefetch(&GI[(size_t)(t + 1) * (3 * DD) + colc + n], 0, 0);
    }
#pragma unroll
    for (int r = 0; r < 8; ++r) {
      int m   = r + mb;
      int col = colc + n;
      float rr = sigm(gir + cr[r] + bhr);
      float zz = sigm(giz + cz[r] + bhz);
      float nn = tanhf(gin + rr * (cn[r] + bhn));
      float hold = hf[m][col];
      bool  msk  = (tcur >= st[m]) && (tcur <= en[m]);
      float hnew = msk ? ((1.f - zz) * nn + zz * hold) : hold;
      hf[m][col]  = hnew;
      hfh[m][col] = (_Float16)hnew;
    }
    if (wave == 0 && lane < MM) {  // local scalar GRU rides along
      bool msk = (tcur >= st[lane]) && (tcur <= en[lane]);
      if (msk) la[lane] = gru1(la[lane], cmb[t], wila, whla, bila, bhla);
    }
  }
  __syncthreads();
  for (int idx = tid; idx < MM * DD; idx += 256) hf_out[idx] = hf[idx >> 7][idx & 127];
  if (tid < MM) la_out[tid] = la[tid];
}

// ---------------------------------------------------------------------------
// K9: refinement (L=3) + heads + distill + final output assembly (114 f32).
// ---------------------------------------------------------------------------
__global__ __launch_bounds__(256) void heads_kernel(
    const float* __restrict__ hf, const float* __restrict__ la_in,
    const float* __restrict__ starts, const float* __restrict__ ends,
    const float* __restrict__ audio_len, const float* __restrict__ Wr1,
    const float* __restrict__ br1, const float* __restrict__ Wr2,
    const float* __restrict__ br2, const float* __restrict__ wparams,
    const float* __restrict__ Wc, const float* __restrict__ bc,
    const float* __restrict__ Wk, const float* __restrict__ bk,
    const float* __restrict__ divv, float* __restrict__ logits_ws,
    float* __restrict__ out) {
  __shared__ float q[MM][DD + 5];
  __shared__ float h1[MM][256];
  __shared__ float s[MM], e[MM], dred[MM];
  int tid = threadIdx.x;
  if (tid < MM) { s[tid] = starts[tid]; e[tid] = ends[tid]; }
  __syncthreads();
  for (int l = 0; l < LL; ++l) {
    if (tid < MM) {
      int m = tid;
      float cc = (s[m] + e[m]) * 0.5f, ww = e[m] - s[m];
      for (int k = 0; k < DD; ++k) q[m][k] = hf[m * DD + k];
      q[m][DD] = cc; q[m][DD + 1] = ww; q[m][DD + 2] = s[m]; q[m][DD + 3] = e[m];
      q[m][DD + 4] = la_in[m];
    }
    __syncthreads();
    for (int idx = tid; idx < MM * 256; idx += 256) {
      int m = idx >> 8, j = idx & 255;
      const float* wr = &Wr1[((size_t)l * 256 + j) * (DD + 5)];
      float a = br1[l * 256 + j];
      for (int k = 0; k < DD + 5; ++k) a += wr[k] * q[m][k];
      h1[m][j] = a > 0.f ? a : 0.f;
    }
    __syncthreads();
    for (int idx = tid; idx < MM * 2 * BB; idx += 256) {
      int m = idx / (2 * BB), o = idx % (2 * BB);
      const float* wr = &Wr2[((size_t)l * 2 * BB + o) * 256];
      float a = br2[l * 2 * BB + o];
      for (int j = 0; j < 256; ++j) a += wr[j] * h1[m][j];
      logits_ws[((size_t)l * MM + m) * (2 * BB) + o] = a;
    }
    __syncthreads();
    if (tid < MM) {
      int m = tid;
      const float* lg = &logits_ws[((size_t)l * MM + m) * (2 * BB)];
      float mx = -1e30f;
      for (int b = 0; b < BB; ++b) mx = fmaxf(mx, lg[b]);
      float se = 0.f;
      for (int b = 0; b < BB; ++b) se += expf(lg[b] - mx);
      float ds = 0.f;
      for (int b = 0; b < BB; ++b) ds += expf(lg[b] - mx) / se * wparams[b];
      float mx2 = -1e30f;
      for (int b = 0; b < BB; ++b) mx2 = fmaxf(mx2, lg[BB + b]);
      float se2 = 0.f;
      for (int b = 0; b < BB; ++b) se2 += expf(lg[BB + b] - mx2);
      float de = 0.f;
      for (int b = 0; b < BB; ++b) de += expf(lg[BB + b] - mx2) / se2 * wparams[b];
      s[m] = fminf(fmaxf(s[m] + ds, 0.f), 1.f);
      e[m] = fminf(fmaxf(e[m] + de, 0.f), 1.f);
    }
    __syncthreads();
  }
  float al = audio_len[0];
  if (tid < MM) { out[2 * tid] = s[tid] * al; out[2 * tid + 1] = e[tid] * al; }
  if (tid < MM) {  // conf
    int m = tid;
    float a = bc[0];
    for (int k = 0; k < DD; ++k) a += Wc[k] * hf[m * DD + k];
    a += Wc[DD] * la_in[m];
    out[2 * MM + m] = a;
  }
  if (tid >= 32 && tid < 32 + MM * (CC - 1)) {  // cls (16 x 4)
    int idx = tid - 32;
    int m = idx >> 2, o = idx & 3;
    const float* wr = &Wk[o * (DD + 1)];
    float a = bk[o];
    for (int k = 0; k < DD; ++k) a += wr[k] * hf[m * DD + k];
    a += wr[DD] * la_in[m];
    out[3 * MM + m * (CC - 1) + o] = a;
  }
  if (tid == 0) out[3 * MM + MM * (CC - 1)] = divv[0];
  // distillation term
  if (tid < MM) {
    int m = tid;
    float dsum = 0.f;
    const float* Llast = &logits_ws[((size_t)(LL - 1) * MM + m) * (2 * BB)];
    for (int half = 0; half < 2; ++half) {
      const float* lt = &Llast[half * BB];
      float mx = -1e30f;
      for (int b = 0; b < BB; ++b) mx = fmaxf(mx, lt[b]);
      float se = 0.f;
      for (int b = 0; b < BB; ++b) se += expf(lt[b] - mx);
      float lse = logf(se) + mx;
      for (int l = 0; l < LL; ++l) {
        const float* ll = &logits_ws[((size_t)l * MM + m) * (2 * BB) + half * BB];
        float mx2 = -1e30f;
        for (int b = 0; b < BB; ++b) mx2 = fmaxf(mx2, ll[b]);
        float se2 = 0.f;
        for (int b = 0; b < BB; ++b) se2 += expf(ll[b] - mx2);
        float lse2 = logf(se2) + mx2;
        float acc = 0.f;
        for (int b = 0; b < BB; ++b) {
          float pt = expf(lt[b] - lse);
          acc += pt * ((lt[b] - lse) - (ll[b] - lse2));
        }
        dsum += acc;
      }
    }
    dred[m] = dsum;
  }
  __syncthreads();
  if (tid == 0) {
    float dtot = 0.f;
    for (int m = 0; m < MM; ++m) dtot += dred[m];
    out[3 * MM + MM * (CC - 1) + 1] = dtot * (1.f / BB);
  }
}

// ---------------------------------------------------------------------------
// Launch
// ---------------------------------------------------------------------------
extern "C" void kernel_launch(void* const* d_in, const int* in_sizes, int n_in,
                              void* d_out, int out_size, void* d_ws, size_t ws_size,
                              hipStream_t stream) {
  const float* emb    = (const float*)d_in[0];
  const float* tpos   = (const float*)d_in[1];
  const float* pred   = (const float*)d_in[2];
  const float* vad    = (const float*)d_in[3];
  const float* alen   = (const float*)d_in[4];
  const float* W_te   = (const float*)d_in[5];
  const float* b_te   = (const float*)d_in[6];
  const float* ln_g   = (const float*)d_in[7];
  const float* ln_b   = (const float*)d_in[8];
  const float* gk     = (const float*)d_in[9];
  const float* wih_ga = (const float*)d_in[10];
  const float* whh_ga = (const float*)d_in[11];
  const float* bih_ga = (const float*)d_in[12];
  const float* bhh_ga = (const float*)d_in[13];
  const float* wih_gv = (const float*)d_in[14];
  const float* whh_gv = (const float*)d_in[15];
  const float* bih_gv = (const float*)d_in[16];
  const float* bhh_gv = (const float*)d_in[17];
  const float* iq     = (const float*)d_in[18];
  const float* Wg1    = (const float*)d_in[19];
  const float* bg1    = (const float*)d_in[20];
  const float* Wg2    = (const float*)d_in[21];
  const float* bg2    = (const float*)d_in[22];
  const float* wih_lf = (const float*)d_in[23];
  const float* whh_lf = (const float*)d_in[24];
  const float* bih_lf = (const float*)d_in[25];
  const float* bhh_lf = (const float*)d_in[26];
  const float* wih_la = (const float*)d_in[27];
  const float* whh_la = (const float*)d_in[28];
  const float* bih_la = (const float*)d_in[29];
  const float* bhh_la = (const float*)d_in[30];
  const float* Wr1    = (const float*)d_in[31];
  const float* br1    = (const float*)d_in[32];
  const float* Wr2    = (const float*)d_in[33];
  const float* br2    = (const float*)d_in[34];
  const float* wpar   = (const float*)d_in[35];
  const float* Wc     = (const float*)d_in[36];
  const float* bc     = (const float*)d_in[37];
  const float* Wk     = (const float*)d_in[38];
  const float* bk     = (const float*)d_in[39];

  char* ws = (char*)d_ws;
  float*    X    = (float*)(ws + OFF_X);
  _Float16* Xh   = (_Float16*)(ws + OFF_XH);
  _Float16* EMBh = (_Float16*)(ws + OFF_EMBH);
  float*    GI   = (float*)(ws + OFF_GI);
  float*    S    = (float*)(ws + OFF_S);
  _Float16* Sh   = (_Float16*)(ws + OFF_SH);
  float*    abn  = (float*)(ws + OFF_ABN);
  float*    vab  = (float*)(ws + OFF_VAB);
  float*    cmb  = (float*)(ws + OFF_CMB);
  float*    qf   = (float*)(ws + OFF_QF);
  float*    st   = (float*)(ws + OFF_ST);
  float*    en   = (float*)(ws + OFF_EN);
  float*    gg   = (float*)(ws + OFF_GG);
  float*    dv   = (float*)(ws + OFF_DIV);
  float*    hff  = (float*)(ws + OFF_HFF);
  float*    la   = (float*)(ws + OFF_LA);
  float*    lgw  = (float*)(ws + OFF_LOG);
  float*    out  = (float*)d_out;

  pre_kernel<<<NN / 256, 256, 0, stream>>>(emb, tpos, pred, vad, W_te, b_te, ln_g, ln_b, gk,
                                           X, Xh, EMBh, abn, vab, cmb);
  ggru_kernel<<<1, 64, 0, stream>>>(abn, vab, wih_ga, whh_ga, bih_ga, bhh_ga,
                                    wih_gv, whh_gv, bih_gv, bhh_gv, gg);
  gi_kernel<<<768, 256, 0, stream>>>(Xh, wih_lf, bih_lf, GI);
  attn_logits_kernel<<<32, 256, 0, stream>>>(iq, EMBh, S);
  softmax_kernel<<<MM, 256, 0, stream>>>(S, Sh);
  qf_kernel<<<1, 256, 0, stream>>>(Sh, Xh, qf);
  interval_kernel<<<1, 256, 0, stream>>>(qf, gg, Wg1, bg1, Wg2, bg2, st, en, dv);
  scan_kernel<<<1, 256, 0, stream>>>(GI, whh_lf, bhh_lf, tpos, cmb, st, en,
                                     wih_la, whh_la, bih_la, bhh_la, hff, la);
  heads_kernel<<<1, 256, 0, stream>>>(hff, la, st, en, alen, Wr1, br1, Wr2, br2, wpar,
                                      Wc, bc, Wk, bk, dv, lgw, out);
}
